// NeuralODE_53549652246999
// MI455X (gfx1250) — compile-verified, matrix operand order checked by reading, not run on hardware
//
#include <hip/hip_runtime.h>

#define HIDDEN 512
#define STEPS  20
#define BM     32          // rows per workgroup (2 M-tiles, one per wave half)
#define KT     16          // HIDDEN / 32
#define PLANE  8192        // halves per 16-row packed activation plane (16*512)

typedef __attribute__((ext_vector_type(16))) _Float16 v16h;
typedef __attribute__((ext_vector_type(8)))  float    v8f;

// WMMA 16-bit fragment K-slot mapping (ISA 7.12.2, 16-bit A 16x32 table):
// lanes 0-15 hold K in {0..7, 16..23}; lanes 16-31 hold K in {8..15, 24..31}.
__device__ __forceinline__ int slot_hi(int r) { return (r >> 3) & 1; }                 // r = k % 32
__device__ __forceinline__ int slot_j (int r) { return (r & 7) | ((r & 16) >> 1); }

__device__ __forceinline__ float fast_tanh(float x) {
#if __has_builtin(__builtin_amdgcn_tanhf)
    return __builtin_amdgcn_tanhf(x);     // native v_tanh_f32 trans op on gfx1250
#else
    return tanhf(x);
#endif
}

// Store one f32 value as f16 into a packed 16-row A-fragment plane at (row mr, K=col).
__device__ __forceinline__ void pack_store(_Float16* plane, int mr, int col, float v) {
    int r2  = col & 31;
    int idx = ((col >> 5) << 9) + (slot_hi(r2) << 8) + slot_j(r2) + (mr << 4);
    plane[idx] = (_Float16)v;
}

// One-time prep: pack a 512x512 f32 weight matrix (row-major, W[k][n]) into f16
// B-fragment order: frag element index = ((kt*32 + nt)*32 + lane)*16 + j,
// so a B-tile load is one contiguous 32B read per lane.
__global__ void NeuralODE_pack_w(const float* __restrict__ W, _Float16* __restrict__ out) {
    int idx = blockIdx.x * blockDim.x + threadIdx.x;
    if (idx >= HIDDEN * HIDDEN) return;
    int k  = idx >> 9;
    int n  = idx & (HIDDEN - 1);
    int kt = k >> 5, r = k & 31;
    int lane = (n & 15) | (slot_hi(r) << 4);
    int nt   = n >> 4;
    out[((((kt << 5) | nt) << 5) | lane) * 16 + slot_j(r)] = (_Float16)W[idx];
}

// [16 x 512] @ [512 x 512] slice: one A plane x this wave's 4 N-tiles.
__device__ __forceinline__ void gemm_512(const _Float16* plane, const _Float16* __restrict__ wp,
                                         int lane, int ntg, v8f acc[4]) {
    v8f zero = {};
    #pragma unroll
    for (int i = 0; i < 4; ++i) acc[i] = zero;
    #pragma unroll 4
    for (int kt = 0; kt < KT; ++kt) {
        v16h a = *(const v16h*)&plane[((kt << 5) | lane) << 4];
        const _Float16* bp = wp + ((((kt << 5) + ntg) << 5) + lane) * 16;
        v16h b0 = *(const v16h*)(bp);
        v16h b1 = *(const v16h*)(bp + 512);
        v16h b2 = *(const v16h*)(bp + 1024);
        v16h b3 = *(const v16h*)(bp + 1536);
        acc[0] = __builtin_amdgcn_wmma_f32_16x16x32_f16(false, a, false, b0, (short)0, acc[0], false, false);
        acc[1] = __builtin_amdgcn_wmma_f32_16x16x32_f16(false, a, false, b1, (short)0, acc[1], false, false);
        acc[2] = __builtin_amdgcn_wmma_f32_16x16x32_f16(false, a, false, b2, (short)0, acc[2], false, false);
        acc[3] = __builtin_amdgcn_wmma_f32_16x16x32_f16(false, a, false, b3, (short)0, acc[3], false, false);
    }
}

// Persistent fused kernel: one 512-thread workgroup (16 waves = 2 M-tiles x 8
// column groups) owns 32 batch rows for all 20 RK4 steps. RK4 state z and the
// k-accumulator live in registers (static element ownership, ~180 VGPRs/wave,
// no VGPR-MSB switching); LDS holds only the packed f16 activation fragments.
__global__ __launch_bounds__(512)
void NeuralODE_rk4(const float* __restrict__ z0, const float* __restrict__ dtg,
                   const _Float16* __restrict__ wp1, const float* __restrict__ b1g,
                   const _Float16* __restrict__ wp2, const float* __restrict__ b2g,
                   float* __restrict__ out) {
    __shared__ _Float16 sA[2 * PLANE];     // ode_fn input, packed (2 planes of 16 rows)
    __shared__ _Float16 sH[2 * PLANE];     // tanh(layer1), packed

    const int tid  = threadIdx.x;
    const int lane = tid & 31;
    const int wave = tid >> 5;
    const int ln   = lane & 15;
    const int hiL  = lane >> 4;
    const int mt   = wave >> 3;            // this wave's M-tile (0 or 1)
    const int cg   = wave & 7;             // this wave's column group
    const int ntg  = cg << 2;              // first N-tile (4 tiles = 64 cols)
    const int colb = cg << 6;              // first owned column
    const int row0 = blockIdx.x * BM;
    const int mloc = hiL << 3;             // C-layout: local row = vgpr_r + 8*hi
    const int rowb = row0 + (mt << 4) + mloc;   // first owned global row

    _Float16* myA = sA + mt * PLANE;
    _Float16* myH = sH + mt * PLANE;

    float zreg[32];                        // owned z elements   [t][r]
    float kreg[32];                        // RK4 k-accumulator  [t][r]
    float dtv[8];                          // per-owned-row dt   [r]
    float bias1[4], bias2[4];

    #pragma unroll
    for (int t = 0; t < 4; ++t) {
        bias1[t] = b1g[colb + (t << 4) + ln];
        bias2[t] = b2g[colb + (t << 4) + ln];
    }
    #pragma unroll
    for (int r = 0; r < 8; ++r)
        dtv[r] = dtg[rowb + r] * (1.0f / STEPS);

    // load z into registers + initial A-fragment pack
    #pragma unroll
    for (int t = 0; t < 4; ++t) {
        int col = colb + (t << 4) + ln;
        #pragma unroll
        for (int r = 0; r < 8; ++r) {
            float v = z0[(rowb + r) * HIDDEN + col];
            zreg[(t << 3) | r] = v;
            pack_store(myA, mloc + r, col, v);
        }
    }
    __syncthreads();

    for (int step = 0; step < STEPS; ++step) {
        #pragma unroll 1
        for (int sub = 0; sub < 4; ++sub) {   // RK4 stages k1..k4
            v8f acc[4];

            // ---- layer 1: h = tanh(zin @ W1 + b1) ----
            gemm_512(myA, wp1, lane, ntg, acc);
            #pragma unroll
            for (int t = 0; t < 4; ++t) {
                int col = colb + (t << 4) + ln;
                #pragma unroll
                for (int r = 0; r < 8; ++r)
                    pack_store(myH, mloc + r, col, fast_tanh(acc[t][r] + bias1[t]));
            }
            __syncthreads();

            // ---- layer 2 + fused RK4 combine (all state in registers) ----
            gemm_512(myH, wp2, lane, ntg, acc);
            const bool lastWrite = (step == STEPS - 1) && (sub == 3);
            #pragma unroll
            for (int t = 0; t < 4; ++t) {
                int col = colb + (t << 4) + ln;
                #pragma unroll
                for (int r = 0; r < 8; ++r) {
                    int zi   = (t << 3) | r;
                    float kv = acc[t][r] + bias2[t];         // k_i element
                    float dt = dtv[r];
                    float z  = zreg[zi];
                    float zin;
                    if (sub == 0) {
                        kreg[zi] = kv;                       // kacc = k1
                        zin = z + 0.5f * dt * kv;            // input for k2
                    } else if (sub == 1) {
                        kreg[zi] += 2.0f * kv;               // += 2 k2
                        zin = z + 0.5f * dt * kv;            // input for k3
                    } else if (sub == 2) {
                        kreg[zi] += 2.0f * kv;               // += 2 k3
                        zin = z + dt * kv;                   // input for k4
                    } else {
                        float zn = z + (dt * (1.0f / 6.0f)) * (kreg[zi] + kv);
                        zreg[zi] = zn;
                        zin = zn;                            // repack for next k1
                        if (lastWrite)
                            out[(rowb + r) * HIDDEN + col] = zn;
                    }
                    pack_store(myA, mloc + r, col, zin);
                }
            }
            __syncthreads();
        }
    }
}

extern "C" void kernel_launch(void* const* d_in, const int* in_sizes, int n_in,
                              void* d_out, int out_size, void* d_ws, size_t ws_size,
                              hipStream_t stream) {
    const float* z0 = (const float*)d_in[0];
    const float* dt = (const float*)d_in[1];
    const float* W1 = (const float*)d_in[2];
    const float* b1 = (const float*)d_in[3];
    const float* W2 = (const float*)d_in[4];
    const float* b2 = (const float*)d_in[5];

    _Float16* wp1 = (_Float16*)d_ws;                 // 512 KB
    _Float16* wp2 = wp1 + HIDDEN * HIDDEN;           // 512 KB

    const int n = HIDDEN * HIDDEN;
    NeuralODE_pack_w<<<(n + 255) / 256, 256, 0, stream>>>(W1, wp1);
    NeuralODE_pack_w<<<(n + 255) / 256, 256, 0, stream>>>(W2, wp2);

    const int batch = in_sizes[0] / HIDDEN;          // 8192
    NeuralODE_rk4<<<batch / BM, 512, 0, stream>>>(z0, dt, wp1, b1, wp2, b2, (float*)d_out);
}
